// FourierAttention_52828097740886
// MI455X (gfx1250) — matrix operand order repaired and checked by reading, not compiled, not run on hardware
//
#include <hip/hip_runtime.h>

// ---------------------------------------------------------------------------
// FourierAttention: rfft->irfft is the identity on real input, so the kernel
// is Y[m,n] = sum_k X[m,k]*W[n,k] + b[n]  with M=B*L=32768, K=N=D=1024.
// Strategy: fp16 WMMA (v_wmma_f32_16x16x32_f16) with f32 accumulation.
//  - pass 0: W fp32 -> fp16 in d_ws (2MB, stays in the 192MB L2)
//  - pass 1: each block = 64 rows x all 1024 cols (32 waves), X tile staged
//            fp32->fp16 in LDS once => X read from HBM exactly once, and W
//            read from L2 only M/64 = 512 times (1GB total L2 traffic).
// ---------------------------------------------------------------------------

typedef __attribute__((ext_vector_type(16))) _Float16 v16h;
typedef __attribute__((ext_vector_type(8)))  _Float16 v8h;
typedef __attribute__((ext_vector_type(4)))  _Float16 v4h;
typedef __attribute__((ext_vector_type(8)))  float    v8f;
typedef __attribute__((ext_vector_type(4)))  float    v4f;

constexpr int BATCH  = 8;
constexpr int SEQ    = 4096;
constexpr int DMODEL = 1024;
constexpr int M = BATCH * SEQ;   // 32768 output rows
constexpr int K = DMODEL;        // 1024 reduction
constexpr int N = DMODEL;        // 1024 output cols
constexpr int MTILE  = 64;       // rows per block (4 row-groups of 16)
constexpr int LDSROW = K + 16;   // halves; +32B pad per row vs 64-bank LDS

// ---- pass 0: convert W [N,K] fp32 -> fp16 (same row-major layout) ----------
__global__ __launch_bounds__(256)
void wconv_kernel(const float* __restrict__ W, _Float16* __restrict__ Wh) {
    int i = (blockIdx.x * 256 + threadIdx.x) * 4;
    v4f w = *(const v4f*)(W + i);
    v4h h = { (_Float16)w.x, (_Float16)w.y, (_Float16)w.z, (_Float16)w.w };
    *(v4h*)(Wh + i) = h;
}

// ---- pass 1: GEMM + bias ---------------------------------------------------
// block: 1024 threads = 32 waves; block handles rows [64*blockIdx.x, +64).
// wave w: row-group (w>>3)*16, columns [128*(w&7), +128) as 8 WMMA tiles.
__global__ __launch_bounds__(1024)
void gemm_kernel(const float* __restrict__ X,
                 const _Float16* __restrict__ Wh,
                 const float* __restrict__ bias,
                 float* __restrict__ Y) {
    __shared__ _Float16 xs[MTILE * LDSROW];   // 64 x 1024 fp16 X tile (+pad)

    const int tid    = threadIdx.x;
    const int m_base = blockIdx.x * MTILE;

    // ---- stage X tile: fp32 global -> fp16 LDS (coalesced, 4 elems/thread) -
    #pragma unroll
    for (int it = 0; it < (MTILE * K) / (1024 * 4); ++it) {   // 16 iters
        int idx = (it * 1024 + tid) * 4;         // 0..65535, step 4
        int row = idx >> 10;                     // /1024
        int col = idx & (K - 1);
        v4f xv = *(const v4f*)(X + (size_t)(m_base + row) * K + col);
        v4h hv = { (_Float16)xv.x, (_Float16)xv.y, (_Float16)xv.z, (_Float16)xv.w };
        *(v4h*)(&xs[row * LDSROW + col]) = hv;
    }
    __syncthreads();

    const int wave    = tid >> 5;        // 0..31
    const int lane    = tid & 31;
    const int l15     = lane & 15;
    const int halfsel = lane >> 4;       // 0: lanes 0-15, 1: lanes 16-31
    const int mgroup  = (wave >> 3) * 16;  // row-group within the block tile
    const int nbase   = (wave & 7) * 128;  // column slice

    v8f acc[8];
    #pragma unroll
    for (int t = 0; t < 8; ++t) acc[t] = v8f{};

    // A fragment source row in LDS (row = l15 for both lane halves)
    const _Float16* xrow = &xs[(mgroup + l15) * LDSROW];

    for (int k = 0; k < K; k += 32) {
        // A (contiguous k order, same as B): lane<16 holds K k..k+15,
        //                          lane>=16 holds K k+16..k+31.
        int ka = k + halfsel * 16;
        v8h a_lo = *(const v8h*)(xrow + ka);        // 16B ds_load
        v8h a_hi = *(const v8h*)(xrow + ka + 8);    // 16B ds_load
        v16h a = __builtin_shufflevector(a_lo, a_hi,
                 0,1,2,3,4,5,6,7,8,9,10,11,12,13,14,15);

        #pragma unroll
        for (int t = 0; t < 8; ++t) {
            // B (32x16): lane n<16 holds W[n, k..k+15], lane n+16 holds
            // W[n, k+16..k+31] -> 32 contiguous bytes per lane, coalesced.
            int col = nbase + t * 16 + l15;
            const _Float16* wp = Wh + (size_t)col * K + k + halfsel * 16;
            v16h bfrag = *(const v16h*)wp;
            acc[t] = __builtin_amdgcn_wmma_f32_16x16x32_f16(
                         /*neg_a=*/false, a, /*neg_b=*/false, bfrag,
                         /*c_mod=*/(short)0, acc[t],
                         /*reuse_a=*/false, /*reuse_b=*/false);
        }
    }

    // ---- bias + store; C/D layout: VGPR r, lanes 0-15 -> M=r, lanes 16-31 ->
    // M=r+8, N = lane&15. Each register store = two 64B coalesced segments.
    const int mrow = m_base + mgroup + halfsel * 8;
    #pragma unroll
    for (int t = 0; t < 8; ++t) {
        int n = nbase + t * 16 + l15;
        float bv = bias[n];
        #pragma unroll
        for (int r = 0; r < 8; ++r) {
            Y[(size_t)(mrow + r) * N + n] = acc[t][r] + bv;
        }
    }
}

extern "C" void kernel_launch(void* const* d_in, const int* in_sizes, int n_in,
                              void* d_out, int out_size, void* d_ws, size_t ws_size,
                              hipStream_t stream) {
    const float* x  = (const float*)d_in[0];   // [B, L, D] fp32
    const float* W  = (const float*)d_in[1];   // [D, D]    fp32 (out, in)
    const float* b  = (const float*)d_in[2];   // [D]       fp32
    float*       y  = (float*)d_out;           // [B, L, D] fp32
    _Float16*    Wh = (_Float16*)d_ws;         // 2MB fp16 copy of W

    // pass 0: N*K / (256 threads * 4 elems) = 1024 blocks
    wconv_kernel<<<(N * K) / (256 * 4), 256, 0, stream>>>(W, Wh);
    // pass 1: one block per 64 output rows
    gemm_kernel<<<M / MTILE, 1024, 0, stream>>>(x, Wh, b, y);
}
